// attention_43516608643187
// MI455X (gfx1250) — compile-verified, matrix operand order checked by reading, not run on hardware
//
#include <hip/hip_runtime.h>
#include <hip/hip_bf16.h>
#include <math.h>

// ---------------------------------------------------------------------------
// MHA block for MI455X (gfx1250, wave32, WMMA f32_16x16x32_f16 + TDM)
//   B=2, S=2048, D=1024, H=16, hd=64
//   kernel 1: qkv = x @ w_qkv              (f32 in -> f16 workspace)
//   kernel 2: flash attention per (b,h)    (f16 -> f16 workspace, K via TDM)
//   kernel 3: out = attn @ w_proj + b      (f16 x f32 -> f32 out)
// Fragments load as ds_load_b128 pairs; GEMM dims are template constants;
// LDS pitches 16B-aligned and bank-conflict-free (80B / 144B).  The attention
// K-tile is DMA'd by the Tensor Data Mover (tensor_load_to_lds) with LDS
// padding reproducing the 144B pitch, synchronized with s_wait_tensorcnt.
// ---------------------------------------------------------------------------

typedef __attribute__((ext_vector_type(16))) _Float16 v16h;
typedef __attribute__((ext_vector_type(8)))  _Float16 v8h;
typedef __attribute__((ext_vector_type(8)))  float    v8f;
typedef __attribute__((ext_vector_type(4)))  unsigned int u32x4;
typedef __attribute__((ext_vector_type(8)))  unsigned int u32x8;

union V16 { v16h v; v8h h[2]; };

#define WMMA_F16(a, b, c) \
    __builtin_amdgcn_wmma_f32_16x16x32_f16(false, (a), false, (b), (short)0, (c), false, false)

// 16-bit 16x32 fragment for lane-group g (= lane>>4): two contiguous 8-half
// runs [g*8 .. g*8+7] and [16+g*8 .. 16+g*8+7] of the lane's row (ISA 7.12.2).
__device__ __forceinline__ v16h load_frag(const _Float16* base, int g) {
    V16 u;
    u.h[0] = *(const v8h*)(base + g * 8);
    u.h[1] = *(const v8h*)(base + 16 + g * 8);
    return u.v;
}

// ---------------------------------------------------------------------------
// Kernel 1: C_f16[M,N] = A_f32[M,K] @ B_f32[K,N]   (N, K compile-time)
// Block tile 128x64, 4 waves (128 thr); wave tile 32x64 (8 WMMAs / k-step).
// ---------------------------------------------------------------------------
template<int N, int K>
__global__ __launch_bounds__(128)
void gemm_f32f32_to_f16(const float* __restrict__ A,
                        const float* __restrict__ B,
                        _Float16* __restrict__ C)
{
    __shared__ alignas(16) _Float16 sA [128][40];  // row-major [m][k]
    __shared__ alignas(16) _Float16 sBt[64][40];   // transposed [n][k]

    const int tid  = threadIdx.x;
    const int lane = tid & 31;
    const int wave = tid >> 5;
    const int lm   = lane & 15;
    const int lg   = lane >> 4;
    const int m0   = blockIdx.y * 128;
    const int n0   = blockIdx.x * 64;

    const int ar = tid >> 2, ag = tid & 3;    // A staging: row base, 8-col group
    const int br = tid >> 4, bg = tid & 15;   // B staging: k-row base, 4-col group

    const float* aPtr = A + (size_t)(m0 + ar) * K + ag * 8;
    const float* bPtr = B + (size_t)br * N + n0 + bg * 4;

    v8f acc[2][4] = {{v8f{}, v8f{}, v8f{}, v8f{}}, {v8f{}, v8f{}, v8f{}, v8f{}}};

    for (int k0 = 0; k0 < K; k0 += 32, aPtr += 32, bPtr += (size_t)32 * N) {
        #pragma unroll
        for (int it = 0; it < 4; ++it) {
            const float* src = aPtr + (size_t)(it * 32) * K;
            v8h o;
            #pragma unroll
            for (int e = 0; e < 8; ++e) o[e] = (_Float16)src[e];
            *(v8h*)&sA[ar + it * 32][ag * 8] = o;
            __builtin_prefetch(src + 32, 0, 3);
        }
        #pragma unroll
        for (int it = 0; it < 4; ++it) {
            const float* src = bPtr + (size_t)(it * 8) * N;
            #pragma unroll
            for (int e = 0; e < 4; ++e)
                sBt[bg * 4 + e][br + it * 8] = (_Float16)src[e];
            __builtin_prefetch(src + (size_t)32 * N, 0, 3);
        }
        __syncthreads();

        v16h af0 = load_frag(&sA[wave * 32 + lm][0],      lg);
        v16h af1 = load_frag(&sA[wave * 32 + 16 + lm][0], lg);
        #pragma unroll
        for (int j = 0; j < 4; ++j) {
            v16h bf = load_frag(&sBt[j * 16 + lm][0], lg);
            acc[0][j] = WMMA_F16(af0, bf, acc[0][j]);
            acc[1][j] = WMMA_F16(af1, bf, acc[1][j]);
        }
        __syncthreads();
    }

    _Float16* cbase = C + (size_t)(m0 + wave * 32 + lg * 8) * N + n0 + lm;
    #pragma unroll
    for (int r2 = 0; r2 < 2; ++r2)
        #pragma unroll
        for (int i = 0; i < 8; ++i) {
            _Float16* cp = cbase + (size_t)(r2 * 16 + i) * N;
            #pragma unroll
            for (int j = 0; j < 4; ++j)
                cp[j * 16] = (_Float16)acc[r2][j][i];
        }
}

// ---------------------------------------------------------------------------
// Kernel 2: flash attention, 64-key chunks (16 WMMAs per barrier round).
// qkv_f16 row = b*2048+s (stride 3072): [q|k|v], head slice at h*64.
// K tile staged by the Tensor Data Mover; V transposed manually.
// ---------------------------------------------------------------------------
__global__ __launch_bounds__(128)
void flash_attn_f16(const _Float16* __restrict__ qkv,
                    _Float16* __restrict__ attn_out)
{
    constexpr int S = 2048, QS = 3072, D = 1024;

    __shared__ alignas(16) _Float16 sK [64][72];      // row-major [key][hd], TDM-filled
    __shared__ alignas(16) _Float16 sVt[64][72];      // transposed [hd][key]
    __shared__ alignas(16) _Float16 sP [4][16][72];   // per-wave [q][key]

    const int tid  = threadIdx.x;
    const int lane = tid & 31;
    const int wave = tid >> 5;
    const int lm   = lane & 15;
    const int lg   = lane >> 4;

    const int bh = blockIdx.y;          // b*16 + h
    const int b  = bh >> 4;
    const int h  = bh & 15;
    const int q0 = blockIdx.x * 64 + wave * 16;

    const _Float16* qbase = qkv + (size_t)b * S * QS + h * 64;

    const int sr = tid >> 3, sg = tid & 7;   // V staging: key-row base, 8-col group
    const _Float16* vPtr = qbase + 2048 + (size_t)sr * QS + sg * 8;

    // --- TDM descriptor for the K tile: 64 rows x 64 f16, global pitch 3072
    //     elements, LDS pitch padded 128B->144B (pad 4 DW every 32 DW). ---
    const unsigned long long kGlobal0 =
        (unsigned long long)(uintptr_t)(qbase + 1024);            // (key 0, hd 0)
    const unsigned ldsK = (unsigned)(uintptr_t)(&sK[0][0]);       // LDS byte address
    u32x8 g1;
    g1[0] = (1u << 16)        // data_size = 2 bytes
          | (1u << 20)        // pad_enable
          | (4u << 22)        // pad_interval: 32 DWORDs
          | (3u << 25);       // pad_amount:   4 DWORDs
    g1[1] = (64u << 16);      // tensor_dim0 = 64 (lo16)
    g1[2] = (2048u << 16);    // tensor_dim0 hi=0 | tensor_dim1 = 2048 (lo16)
    g1[3] = (64u << 16);      // tensor_dim1 hi=0 | tile_dim0 = 64
    g1[4] = 64u;              // tile_dim1 = 64, tile_dim2 = 0
    g1[5] = 3072u;            // tensor_dim0_stride lo32
    g1[6] = 0u;
    g1[7] = 0u;

    // Q fragments (16 rows x 64 hd): 4x global b128 loads, resident all loop
    const _Float16* qrow = qbase + (size_t)(q0 + lm) * QS + lg * 8;
    V16 uq0, uq1;
    uq0.h[0] = *(const v8h*)(qrow);
    uq0.h[1] = *(const v8h*)(qrow + 16);
    uq1.h[0] = *(const v8h*)(qrow + 32);
    uq1.h[1] = *(const v8h*)(qrow + 48);
    const v16h qf0 = uq0.v, qf1 = uq1.v;

    v8f acc_o[4] = {v8f{}, v8f{}, v8f{}, v8f{}};
    float rowmax[8], rowsum[8];
    #pragma unroll
    for (int i = 0; i < 8; ++i) { rowmax[i] = -INFINITY; rowsum[i] = 0.0f; }

    for (int kc = 0; kc < S; kc += 64, vPtr += (size_t)64 * QS) {
        // wave 0 kicks the TDM for the K tile (EXEC-independent DMA)
        if (wave == 0) {
            unsigned long long ga = kGlobal0 + (unsigned long long)kc * (QS * 2);
            u32x4 g0;
            g0[0] = 1u;                                   // count = 1 descriptor
            g0[1] = ldsK;                                 // LDS destination
            g0[2] = (unsigned)ga;                         // global addr lo32
            g0[3] = ((unsigned)(ga >> 32) & 0x01FFFFFFu)  // global addr [56:32]
                  | 0x80000000u;                          // type = 2 ("image")
            asm volatile("tensor_load_to_lds %0, %1" :: "s"(g0), "s"(g1) : "memory");
        }
        // meanwhile all waves stage V (transposed into [hd][key])
        #pragma unroll
        for (int it = 0; it < 4; ++it) {
            const _Float16* vs = vPtr + (size_t)(it * 16) * QS;
            v8h vv = *(const v8h*)vs;
            #pragma unroll
            for (int e = 0; e < 8; ++e) sVt[sg * 8 + e][sr + it * 16] = vv[e];
            __builtin_prefetch(vs + (size_t)64 * QS, 0, 3);
        }
        if (wave == 0) __builtin_amdgcn_s_wait_tensorcnt(0);
        __syncthreads();

        // scores S_j (16q x 16k), j=0..3 : Q(16x64) @ K_j^T
        v8f sc[4] = {v8f{}, v8f{}, v8f{}, v8f{}};
        #pragma unroll
        for (int j = 0; j < 4; ++j) {
            const _Float16* krow = &sK[j * 16 + lm][0];
            v16h bf0 = load_frag(krow,      lg);   // hd 0..31
            v16h bf1 = load_frag(krow + 32, lg);   // hd 32..63
            sc[j] = WMMA_F16(qf0, bf0, sc[j]);
            sc[j] = WMMA_F16(qf1, bf1, sc[j]);
        }

        // online softmax in D-layout: register i = query row (m = i + 8*lg),
        // lane lm = key (+16j).  Row reduction via shfl_xor over half-wave.
        #pragma unroll
        for (int i = 0; i < 8; ++i) {
            float s[4];
            #pragma unroll
            for (int j = 0; j < 4; ++j) s[j] = sc[j][i] * 0.125f;  // 1/sqrt(64)
            float cmax = fmaxf(fmaxf(s[0], s[1]), fmaxf(s[2], s[3]));
            #pragma unroll
            for (int off = 1; off < 16; off <<= 1)
                cmax = fmaxf(cmax, __shfl_xor(cmax, off, 32));
            float mnew  = fmaxf(rowmax[i], cmax);
            float alpha = __expf(rowmax[i] - mnew);
            rowmax[i]   = mnew;
            float p[4], psum = 0.0f;
            #pragma unroll
            for (int j = 0; j < 4; ++j) { p[j] = __expf(s[j] - mnew); psum += p[j]; }
            #pragma unroll
            for (int off = 1; off < 16; off <<= 1)
                psum += __shfl_xor(psum, off, 32);
            rowsum[i] = rowsum[i] * alpha + psum;
            #pragma unroll
            for (int c = 0; c < 4; ++c)
                acc_o[c][i] = acc_o[c][i] * alpha;
            int m = i + lg * 8;
            #pragma unroll
            for (int j = 0; j < 4; ++j)
                sP[wave][m][j * 16 + lm] = (_Float16)p[j];
        }
        __syncthreads();

        // O += P(16x64) @ V(64x64)
        v16h pf0 = load_frag(&sP[wave][lm][0],  lg);   // keys 0..31
        v16h pf1 = load_frag(&sP[wave][lm][32], lg);   // keys 32..63
        #pragma unroll
        for (int c = 0; c < 4; ++c) {
            v16h vf0 = load_frag(&sVt[c * 16 + lm][0],  lg);
            v16h vf1 = load_frag(&sVt[c * 16 + lm][32], lg);
            acc_o[c] = WMMA_F16(pf0, vf0, acc_o[c]);
            acc_o[c] = WMMA_F16(pf1, vf1, acc_o[c]);
        }
        __syncthreads();
    }

    // normalize, write f16 [B,S,D], D-index = h*64 + hd (matches ref transpose)
    _Float16* obase = attn_out + ((size_t)b * S + q0 + lg * 8) * D + h * 64 + lm;
    #pragma unroll
    for (int i = 0; i < 8; ++i) {
        float inv = 1.0f / rowsum[i];
        _Float16* op = obase + (size_t)i * D;
        #pragma unroll
        for (int c = 0; c < 4; ++c)
            op[c * 16] = (_Float16)(acc_o[c][i] * inv);
    }
}

// ---------------------------------------------------------------------------
// Kernel 3: C_f32[M,N] = A_f16[M,K] @ B_f32[K,N] + bias[N]   (N, K compile-time)
// ---------------------------------------------------------------------------
template<int N, int K>
__global__ __launch_bounds__(128)
void gemm_f16f32_bias_f32(const _Float16* __restrict__ A,
                          const float* __restrict__ B,
                          const float* __restrict__ bias,
                          float* __restrict__ C)
{
    __shared__ alignas(16) _Float16 sA [128][40];
    __shared__ alignas(16) _Float16 sBt[64][40];

    const int tid  = threadIdx.x;
    const int lane = tid & 31;
    const int wave = tid >> 5;
    const int lm   = lane & 15;
    const int lg   = lane >> 4;
    const int m0   = blockIdx.y * 128;
    const int n0   = blockIdx.x * 64;

    const int ar = tid >> 2, ag = tid & 3;
    const int br = tid >> 4, bg = tid & 15;

    const _Float16* aPtr = A + (size_t)(m0 + ar) * K + ag * 8;
    const float*    bPtr = B + (size_t)br * N + n0 + bg * 4;

    v8f acc[2][4] = {{v8f{}, v8f{}, v8f{}, v8f{}}, {v8f{}, v8f{}, v8f{}, v8f{}}};

    for (int k0 = 0; k0 < K; k0 += 32, aPtr += 32, bPtr += (size_t)32 * N) {
        #pragma unroll
        for (int it = 0; it < 4; ++it) {
            const _Float16* src = aPtr + (size_t)(it * 32) * K;
            *(v8h*)&sA[ar + it * 32][ag * 8] = *(const v8h*)src;
            __builtin_prefetch(src + 32, 0, 3);
        }
        #pragma unroll
        for (int it = 0; it < 4; ++it) {
            const float* src = bPtr + (size_t)(it * 8) * N;
            #pragma unroll
            for (int e = 0; e < 4; ++e)
                sBt[bg * 4 + e][br + it * 8] = (_Float16)src[e];
            __builtin_prefetch(src + (size_t)32 * N, 0, 3);
        }
        __syncthreads();

        v16h af0 = load_frag(&sA[wave * 32 + lm][0],      lg);
        v16h af1 = load_frag(&sA[wave * 32 + 16 + lm][0], lg);
        #pragma unroll
        for (int j = 0; j < 4; ++j) {
            v16h bf = load_frag(&sBt[j * 16 + lm][0], lg);
            acc[0][j] = WMMA_F16(af0, bf, acc[0][j]);
            acc[1][j] = WMMA_F16(af1, bf, acc[1][j]);
        }
        __syncthreads();
    }

    const float bj[4] = {bias[n0 + lm], bias[n0 + 16 + lm],
                         bias[n0 + 32 + lm], bias[n0 + 48 + lm]};

    float* cbase = C + (size_t)(m0 + wave * 32 + lg * 8) * N + n0 + lm;
    #pragma unroll
    for (int r2 = 0; r2 < 2; ++r2)
        #pragma unroll
        for (int i = 0; i < 8; ++i) {
            float* cp = cbase + (size_t)(r2 * 16 + i) * N;
            #pragma unroll
            for (int j = 0; j < 4; ++j)
                cp[j * 16] = acc[r2][j][i] + bj[j];
        }
}

// ---------------------------------------------------------------------------
extern "C" void kernel_launch(void* const* d_in, const int* in_sizes, int n_in,
                              void* d_out, int out_size, void* d_ws, size_t ws_size,
                              hipStream_t stream)
{
    (void)in_sizes; (void)n_in; (void)out_size; (void)ws_size;

    const float* x      = (const float*)d_in[0];   // [2,2048,1024]
    const float* w_qkv  = (const float*)d_in[1];   // [1024,3072]
    const float* w_proj = (const float*)d_in[2];   // [1024,1024]
    const float* b_proj = (const float*)d_in[3];   // [1024]
    float*       out    = (float*)d_out;           // [2,2048,1024]

    constexpr int M = 2 * 2048;   // flattened B*S rows
    constexpr int D = 1024;

    _Float16* qkv_h  = (_Float16*)d_ws;                       // [4096,3072] f16 (24 MB)
    _Float16* attn_h = qkv_h + (size_t)M * 3 * D;             // [4096,1024] f16 (8 MB)

    // 1) qkv projection: [4096,1024] @ [1024,3072] -> f16
    gemm_f32f32_to_f16<3 * D, D><<<dim3(3 * D / 64, M / 128), 128, 0, stream>>>(
        x, w_qkv, qkv_h);

    // 2) flash attention: grid = (q-tiles, b*H)
    flash_attn_f16<<<dim3(2048 / 64, 2 * 16), 128, 0, stream>>>(qkv_h, attn_h);

    // 3) output projection + bias: [4096,1024] @ [1024,1024] -> f32
    gemm_f16f32_bias_f32<D, D><<<dim3(D / 64, M / 128), 128, 0, stream>>>(
        attn_h, w_proj, b_proj, out);
}
